// QuantLinear_47339129536590
// MI455X (gfx1250) — matrix operand order repaired and checked by reading, not compiled, not run on hardware
//
#include <hip/hip_runtime.h>

typedef __attribute__((ext_vector_type(16))) _Float16 v16h;
typedef __attribute__((ext_vector_type(8)))  _Float16 v8h;
typedef __attribute__((ext_vector_type(8)))  float    v8f;

#define BM 128
#define BN 256
#define BK 32
#define NTHREADS 256
#define AS_STRIDE 40   // halfs per A row in LDS (padded, 16B-aligned offsets)
#define BS_STRIDE 40   // halfs per B column in LDS

// CDNA5 async global->LDS copy (bypasses VGPRs, tracked with ASYNCcnt).
// VDST = per-lane LDS byte address, VADDR = per-lane 64-bit global address.
__device__ __forceinline__ void async_copy_b128(unsigned lds_off,
                                                unsigned long long gaddr) {
    asm volatile("global_load_async_to_lds_b128 %0, %1, off"
                 :: "v"(lds_off), "v"(gaddr) : "memory");
}
__device__ __forceinline__ void wait_asynccnt0() {
    asm volatile("s_wait_asynccnt 0x0" ::: "memory");
}

__global__ __launch_bounds__(NTHREADS)
void gptq_gemm_wmma(const _Float16* __restrict__ x,
                    const unsigned int* __restrict__ qweight,
                    const unsigned int* __restrict__ qzeros,
                    const _Float16* __restrict__ scales,
                    const _Float16* __restrict__ bias,
                    _Float16* __restrict__ out,
                    int M, int N, int K, int G)
{
    __shared__ _Float16 As[BM * AS_STRIDE];   // row-major [m][k]
    __shared__ _Float16 Bs[BN * BS_STRIDE];   // column-major [n][k]

    const int tid  = threadIdx.x;
    const int lane = tid & 31;
    const int wave = tid >> 5;
    const int wm   = wave >> 2;     // 0..1  (M direction, 64 rows each)
    const int wn   = wave & 3;      // 0..3  (N direction, 64 cols each)
    const int hi   = lane >> 4;     // half-wave select
    const int lm   = lane & 15;

    const int bn0 = blockIdx.x * BN;
    const int bm0 = blockIdx.y * BM;

    // A staging role: thread -> (row, two 8-half chunks)
    const int ar = tid >> 1;            // 0..127
    const int ac = (tid & 1) << 4;      // 0 or 16

    // B dequant role: one tile column per thread
    const int n  = tid;                 // 0..255
    const int gn = bn0 + n;             // global column

    const int groupsize = K / G;                 // 128
    const int kt_per_g  = groupsize / BK;        // 4
    const int qz_stride = N >> 3;

    v8f acc[4][4] = {};

    for (int g = 0; g < G; ++g) {
        // per-group dequant constants for this thread's column
        const float sc = (float)scales[(size_t)g * N + gn];
        const unsigned int zp = qzeros[(size_t)g * qz_stride + (gn >> 3)];
        const float zoff = (float)(((zp >> ((gn & 7) * 4)) & 15u) + 1u) * sc;

        for (int kt = 0; kt < kt_per_g; ++kt) {
            const int k0 = (g * kt_per_g + kt) * BK;

            __syncthreads();   // previous iteration's LDS reads complete

            // ---- stage A tile: BM x BK fp16, async global->LDS ----
            {
                const _Float16* src = &x[(size_t)(bm0 + ar) * K + k0 + ac];
                const unsigned  dst =
                    (unsigned)(size_t)&As[ar * AS_STRIDE + ac];
                async_copy_b128(dst,      (unsigned long long)(size_t)src);
                async_copy_b128(dst + 16, (unsigned long long)(size_t)(src + 8));
            }

            // ---- stage B tile: int4 -> fp16 dequant, column-major ----
            const unsigned int* qw = qweight + (size_t)(k0 >> 3) * N + gn;
            #pragma unroll
            for (int kr = 0; kr < 4; ++kr) {
                const unsigned int q = qw[(size_t)kr * N];
                v8h wv;
                #pragma unroll
                for (int j = 0; j < 8; ++j) {
                    const float w =
                        (float)((q >> (4 * j)) & 15u) * sc - zoff;
                    wv[j] = (_Float16)w;
                }
                *(v8h*)&Bs[n * BS_STRIDE + kr * 8] = wv;
            }

            // prefetch next qweight rows into L2 while we compute
            if (k0 + BK < K)
                __builtin_prefetch(qw + (size_t)4 * N, 0, 0);

            wait_asynccnt0();  // async A-tile landed in LDS
            __syncthreads();

            // ---- A fragments (16x32 f16, documented gfx1250 layout) ----
            v16h afrag[4];
            #pragma unroll
            for (int mi = 0; mi < 4; ++mi) {
                const int row = wm * 64 + mi * 16 + lm;
                const v8h lo = *(const v8h*)&As[row * AS_STRIDE + hi * 8];
                const v8h hh = *(const v8h*)&As[row * AS_STRIDE + hi * 8 + 16];
                afrag[mi] = __builtin_shufflevector(lo, hh,
                    0,1,2,3,4,5,6,7,8,9,10,11,12,13,14,15);
            }

            // ---- B fragments (32x16 f16): same K order as A fragments ----
            v16h bfrag[4];
            #pragma unroll
            for (int ni = 0; ni < 4; ++ni) {
                const int col = wn * 64 + ni * 16 + lm;
                const v8h lo = *(const v8h*)&Bs[col * BS_STRIDE + hi * 8];
                const v8h hh = *(const v8h*)&Bs[col * BS_STRIDE + hi * 8 + 16];
                bfrag[ni] = __builtin_shufflevector(lo, hh,
                    0,1,2,3,4,5,6,7,8,9,10,11,12,13,14,15);
            }

            // ---- 16x v_wmma_f32_16x16x32_f16 ----
            #pragma unroll
            for (int mi = 0; mi < 4; ++mi)
                #pragma unroll
                for (int ni = 0; ni < 4; ++ni)
                    acc[mi][ni] = __builtin_amdgcn_wmma_f32_16x16x32_f16(
                        false, afrag[mi], false, bfrag[ni],
                        (short)0, acc[mi][ni], false, false);
        }
    }

    // ---- epilogue: add bias, convert to fp16, store ----
    #pragma unroll
    for (int mi = 0; mi < 4; ++mi) {
        #pragma unroll
        for (int ni = 0; ni < 4; ++ni) {
            const int gcol = bn0 + wn * 64 + ni * 16 + lm;
            const float bv = (float)bias[gcol];
            #pragma unroll
            for (int r = 0; r < 8; ++r) {
                const int grow = bm0 + wm * 64 + mi * 16 + hi * 8 + r;
                out[(size_t)grow * N + gcol] =
                    (_Float16)(acc[mi][ni][r] + bv);
            }
        }
    }
}

extern "C" void kernel_launch(void* const* d_in, const int* in_sizes, int n_in,
                              void* d_out, int out_size, void* d_ws, size_t ws_size,
                              hipStream_t stream) {
    const _Float16*     x       = (const _Float16*)d_in[0];
    const unsigned int* qweight = (const unsigned int*)d_in[1];
    const unsigned int* qzeros  = (const unsigned int*)d_in[2];
    const _Float16*     scales  = (const _Float16*)d_in[3];
    const _Float16*     bias    = (const _Float16*)d_in[4];
    _Float16*           out     = (_Float16*)d_out;

    const int N = in_sizes[4];                   // 11008
    const int K = (in_sizes[1] / N) * 8;         // 4096
    const int M = in_sizes[0] / K;               // 8192
    const int G = in_sizes[3] / N;               // 32

    dim3 grid(N / BN, M / BM);
    dim3 block(NTHREADS);
    gptq_gemm_wmma<<<grid, block, 0, stream>>>(x, qweight, qzeros, scales,
                                               bias, out, M, N, K, G);
}
